// APPNPNet_69002944578215
// MI455X (gfx1250) — matrix operand order repaired and checked by reading, not compiled, not run on hardware
//
#include <hip/hip_runtime.h>

#define ALPHA 0.5f

typedef __attribute__((ext_vector_type(16))) __bf16        v16bf;
typedef __attribute__((ext_vector_type(8)))  float         v8f;
typedef __attribute__((ext_vector_type(8)))  unsigned int  v8u;

// RNE float->bf16 (used only in one-time weight packing).
__device__ __forceinline__ unsigned short f2bf_rne(float f) {
  unsigned int x = __float_as_uint(f);
  x += 0x7FFFu + ((x >> 16) & 1u);
  return (unsigned short)(x >> 16);
}
__device__ __forceinline__ unsigned int pack2bf_rne(float lo, float hi) {
  return (unsigned int)f2bf_rne(lo) | ((unsigned int)f2bf_rne(hi) << 16);
}
// Truncating pack of two floats -> one dword of bf16 pair: single v_perm_b32.
__device__ __forceinline__ unsigned int pack2bf(float f0, float f1) {
  return __builtin_amdgcn_perm(__float_as_uint(f1), __float_as_uint(f0), 0x07060302u);
}

// ---------------- degree / norm ----------------
__global__ void k_deg_init(float* __restrict__ deg, int n) {
  int i = blockIdx.x * blockDim.x + threadIdx.x;
  if (i < n) deg[i] = 1.0f;  // self-loop contribution
}

__global__ void k_deg_edges(const int* __restrict__ dst, float* __restrict__ deg, int e) {
  int i = blockIdx.x * blockDim.x + threadIdx.x;
  if (i < e) atomicAdd(&deg[dst[i]], 1.0f);
}

__global__ void k_rsqrt_inplace(float* __restrict__ d, int n) {
  int i = blockIdx.x * blockDim.x + threadIdx.x;
  if (i < n) d[i] = rsqrtf(d[i]);  // deg >= 1 always (self loop)
}

// ---------------- one-time weight packing into B-fragment layout ----------------
// P1[n*256 + kp] = bf16pair(W1[2kp][n], W1[2kp+1][n]); K padded 500->512
__global__ void k_pack_W1(const float* __restrict__ W1, unsigned int* __restrict__ P1) {
  int t = blockIdx.x * blockDim.x + threadIdx.x;
  if (t >= 64 * 256) return;
  int n  = t >> 8;
  int kp = t & 255;
  int k  = kp * 2;
  float f0 = (k     < 500) ? W1[k * 64 + n]       : 0.0f;
  float f1 = (k + 1 < 500) ? W1[(k + 1) * 64 + n] : 0.0f;
  P1[t] = pack2bf_rne(f0, f1);
}

// P2[n*32 + kp] = bf16pair(W2[2kp][n], W2[2kp+1][n]); n padded 40->48, K=64 exact
__global__ void k_pack_W2(const float* __restrict__ W2, unsigned int* __restrict__ P2) {
  int t = blockIdx.x * blockDim.x + threadIdx.x;
  if (t >= 48 * 32) return;
  int n  = t >> 5;
  int kp = t & 31;
  int k  = kp * 2;
  float f0 = (n < 40) ? W2[k * 40 + n]       : 0.0f;
  float f1 = (n < 40) ? W2[(k + 1) * 40 + n] : 0.0f;
  P2[t] = pack2bf_rne(f0, f1);
}

// ---------------- GEMM1 chunk helper ----------------
template <bool GUARD>
__device__ __forceinline__ void gemm1_chunk(int kc, const float* __restrict__ xr,
                                            const unsigned int* __restrict__ P1,
                                            int ll, int hi, v8f acc[4]) {
  const int k0 = kc * 32;
  // A fragment (contiguous k order, k = k0 + 16*hi + 2d, same as B): lanes=M, dword d holds an even-K pair.
  v8u ap;
#pragma unroll
  for (int d = 0; d < 8; ++d) {
    int kk = k0 + 2 * d + hi * 16;  // always even
    float2 f;
    if (GUARD) f = (kk < 500) ? *(const float2*)(xr + kk) : make_float2(0.0f, 0.0f);
    else       f = *(const float2*)(xr + kk);
    ap[d] = pack2bf(f.x, f.y);
  }
  v16bf av = __builtin_bit_cast(v16bf, ap);

  // Fetch ALL four B fragments first (8 b128 loads in one clause), then 4 WMMAs.
  uint4 q[8];
#pragma unroll
  for (int t = 0; t < 4; ++t) {
    const uint4* bq = (const uint4*)(P1 + ((t * 16 + ll) << 8) + (kc << 4) + (hi << 3));
    q[2 * t]     = bq[0];
    q[2 * t + 1] = bq[1];
  }
#pragma unroll
  for (int t = 0; t < 4; ++t) {
    v8u bp;
    bp[0] = q[2 * t].x;     bp[1] = q[2 * t].y;
    bp[2] = q[2 * t].z;     bp[3] = q[2 * t].w;
    bp[4] = q[2 * t + 1].x; bp[5] = q[2 * t + 1].y;
    bp[6] = q[2 * t + 1].z; bp[7] = q[2 * t + 1].w;
    v16bf bv = __builtin_bit_cast(v16bf, bp);
    acc[t] = __builtin_amdgcn_wmma_f32_16x16x32_bf16(
        false, av, false, bv, (short)0, acc[t], false, false);
  }
}

// ---------------- GEMM1: h0 = x @ W1 + b1 ----------------
__global__ void __launch_bounds__(256)
k_gemm1(const float* __restrict__ x, const unsigned int* __restrict__ P1,
        const float* __restrict__ b1, float* __restrict__ h0,
        float* __restrict__ hcur, int mtiles) {
  const int lane = threadIdx.x & 31;
  const int wave = threadIdx.x >> 5;
  const int tile = blockIdx.x * 8 + wave;   // wave-uniform -> EXEC stays full for WMMA
  if (tile >= mtiles) return;
  const int ll = lane & 15;
  const int hi = (lane >> 4) & 1;
  const int row = tile * 16 + ll;
  const float* xr = x + (long long)row * 500;

  v8f acc[4] = {};
  // Chunks 0..14: max kk = 14*32 + 16 + 6 + 8 + 1 = 479 < 500 -> no guards.
  for (int kc = 0; kc < 15; ++kc)
    gemm1_chunk<false>(kc, xr, P1, ll, hi, acc);
  // Tail chunk 15 covers K 480..511 -> guarded loads, zero-padded.
  gemm1_chunk<true>(15, xr, P1, ll, hi, acc);

#pragma unroll
  for (int t = 0; t < 4; ++t) {
    const int n = t * 16 + ll;
    const float bias = b1[n];
#pragma unroll
    for (int r = 0; r < 8; ++r) {
      const int m = tile * 16 + r + hi * 8;  // C/D layout: VGPR r = row r (lo) / r+8 (hi)
      const float v = acc[t][r] + bias;
      h0[m * 64 + n]   = v;
      hcur[m * 64 + n] = v;
    }
  }
}

// ---------------- propagation step kernels ----------------
__global__ void k_self_init(const float* __restrict__ dinv, const float* __restrict__ hcur,
                            float* __restrict__ acc, int n) {
  int t = blockIdx.x * blockDim.x + threadIdx.x;
  int i = t >> 4, c = t & 15;
  if (i >= n) return;
  float w = dinv[i]; w *= w;  // self-loop norm = dinv[i]^2
  float4 v = ((const float4*)(hcur + (long long)i * 64))[c];
  float4 r = make_float4(w * v.x, w * v.y, w * v.z, w * v.w);
  ((float4*)(acc + (long long)i * 64))[c] = r;
}

__global__ void k_edge_prop(const int* __restrict__ src, const int* __restrict__ dst,
                            const float* __restrict__ dinv, const float* __restrict__ hcur,
                            float* __restrict__ acc, int e) {
  long long t = (long long)blockIdx.x * blockDim.x + threadIdx.x;
  int i = (int)(t >> 4), c = (int)(t & 15);
  if (i >= e) return;
  int s = src[i], d = dst[i];
  float w = dinv[s] * dinv[d];
  float4 v = ((const float4*)(hcur + (long long)s * 64))[c];
  float* p = acc + (long long)d * 64 + c * 4;
  atomicAdd(p + 0, w * v.x);
  atomicAdd(p + 1, w * v.y);
  atomicAdd(p + 2, w * v.z);
  atomicAdd(p + 3, w * v.w);
}

__global__ void k_combine(const float* __restrict__ acc, const float* __restrict__ h0,
                          float* __restrict__ hcur, int n) {
  int t = blockIdx.x * blockDim.x + threadIdx.x;
  int i = t >> 4, c = t & 15;
  if (i >= n) return;
  float4 a = ((const float4*)(acc + (long long)i * 64))[c];
  float4 z = ((const float4*)(h0  + (long long)i * 64))[c];
  float4 r = make_float4((1.0f - ALPHA) * a.x + ALPHA * z.x,
                         (1.0f - ALPHA) * a.y + ALPHA * z.y,
                         (1.0f - ALPHA) * a.z + ALPHA * z.z,
                         (1.0f - ALPHA) * a.w + ALPHA * z.w);
  ((float4*)(hcur + (long long)i * 64))[c] = r;
}

// ---------------- GEMM2: out = relu(h) @ W2 + b2  (N=40 padded to 48) ----------------
__global__ void __launch_bounds__(256)
k_gemm2(const float* __restrict__ h, const unsigned int* __restrict__ P2,
        const float* __restrict__ b2, float* __restrict__ out, int mtiles) {
  const int lane = threadIdx.x & 31;
  const int wave = threadIdx.x >> 5;
  const int tile = blockIdx.x * 8 + wave;
  if (tile >= mtiles) return;
  const int ll = lane & 15;
  const int hi = (lane >> 4) & 1;
  const int row = tile * 16 + ll;
  const float* hr = h + (long long)row * 64;

  v8f acc[3] = {};
#pragma unroll
  for (int kc = 0; kc < 2; ++kc) {
    const int k0 = kc * 32;
    v8u ap;
#pragma unroll
    for (int d = 0; d < 8; ++d) {
      int kk = k0 + 2 * d + hi * 16;  // < 64 always
      float2 f = *(const float2*)(hr + kk);
      ap[d] = pack2bf(fmaxf(f.x, 0.0f), fmaxf(f.y, 0.0f));  // fused ReLU
    }
    v16bf av = __builtin_bit_cast(v16bf, ap);

    uint4 q[6];
#pragma unroll
    for (int t = 0; t < 3; ++t) {
      const uint4* bq = (const uint4*)(P2 + ((t * 16 + ll) << 5) + (kc << 4) + (hi << 3));
      q[2 * t]     = bq[0];
      q[2 * t + 1] = bq[1];
    }
#pragma unroll
    for (int t = 0; t < 3; ++t) {
      v8u bp;
      bp[0] = q[2 * t].x;     bp[1] = q[2 * t].y;
      bp[2] = q[2 * t].z;     bp[3] = q[2 * t].w;
      bp[4] = q[2 * t + 1].x; bp[5] = q[2 * t + 1].y;
      bp[6] = q[2 * t + 1].z; bp[7] = q[2 * t + 1].w;
      v16bf bv = __builtin_bit_cast(v16bf, bp);
      acc[t] = __builtin_amdgcn_wmma_f32_16x16x32_bf16(
          false, av, false, bv, (short)0, acc[t], false, false);
    }
  }
#pragma unroll
  for (int t = 0; t < 3; ++t) {
    const int n = t * 16 + ll;
    if (n < 40) {
      const float bias = b2[n];
#pragma unroll
      for (int r = 0; r < 8; ++r) {
        const int m = tile * 16 + r + hi * 8;
        out[m * 40 + n] = acc[t][r] + bias;
      }
    }
  }
}

// ---------------- host-side orchestration ----------------
extern "C" void kernel_launch(void* const* d_in, const int* in_sizes, int n_in,
                              void* d_out, int out_size, void* d_ws, size_t ws_size,
                              hipStream_t stream) {
  const float* x  = (const float*)d_in[0];
  const int*   ei = (const int*)d_in[1];
  const float* W1 = (const float*)d_in[2];
  const float* b1 = (const float*)d_in[3];
  const float* W2 = (const float*)d_in[4];
  const float* b2 = (const float*)d_in[5];
  float* out = (float*)d_out;

  const int N = in_sizes[0] / 500;   // 100000
  const int E = in_sizes[1] / 2;     // 1600000
  const int* src = ei;
  const int* dst = ei + E;

  char* ws = (char*)d_ws;
  const size_t szH = (size_t)N * 64 * sizeof(float);
  float*        h0   = (float*)(ws);
  float*        hcur = (float*)(ws + szH);
  float*        accb = (float*)(ws + 2 * szH);
  float*        dinv = (float*)(ws + 3 * szH);
  unsigned int* P1   = (unsigned int*)(ws + 3 * szH + (((size_t)N * 4 + 255) & ~(size_t)255));
  unsigned int* P2   = P1 + 64 * 256;

  const int mtiles = N / 16;  // N is a multiple of 16
  dim3 blk(256);

  k_deg_init<<<dim3((N + 255) / 256), blk, 0, stream>>>(dinv, N);
  k_deg_edges<<<dim3((E + 255) / 256), blk, 0, stream>>>(dst, dinv, E);
  k_rsqrt_inplace<<<dim3((N + 255) / 256), blk, 0, stream>>>(dinv, N);

  k_pack_W1<<<dim3((64 * 256 + 255) / 256), blk, 0, stream>>>(W1, P1);
  k_pack_W2<<<dim3((48 * 32 + 255) / 256), blk, 0, stream>>>(W2, P2);

  k_gemm1<<<dim3((mtiles + 7) / 8), blk, 0, stream>>>(x, P1, b1, h0, hcur, mtiles);

  const unsigned nodeBlocks = (unsigned)(((long long)N * 16 + 255) / 256);
  const unsigned edgeBlocks = (unsigned)(((long long)E * 16 + 255) / 256);
  for (int step = 0; step < 5; ++step) {
    k_self_init<<<dim3(nodeBlocks), blk, 0, stream>>>(dinv, hcur, accb, N);
    k_edge_prop<<<dim3(edgeBlocks), blk, 0, stream>>>(src, dst, dinv, hcur, accb, E);
    k_combine<<<dim3(nodeBlocks), blk, 0, stream>>>(accb, h0, hcur, N);
  }

  k_gemm2<<<dim3((mtiles + 7) / 8), blk, 0, stream>>>(hcur, P2, b2, out, mtiles);
}
